// BahdanauAttention_15178414424090
// MI455X (gfx1250) — compile-verified, matrix operand order checked by reading, not run on hardware
//
#include <hip/hip_runtime.h>
#include <stdint.h>

#define B_ 128
#define S_ 1024
#define H_ 512
#define NEGV 1e10f

typedef __attribute__((ext_vector_type(16))) __bf16 v16bf;
typedef __attribute__((ext_vector_type(8)))  float  v8f;

union Frag { uint32_t u[8]; v16bf v; };

__device__ __forceinline__ uint16_t f2bf(float f) {
    uint32_t u = __float_as_uint(f);
    uint32_t r = (u + 0x7FFFu + ((u >> 16) & 1u)) >> 16;   // RNE
    return (uint16_t)r;
}
__device__ __forceinline__ uint32_t pack2bf(float x, float y) {
    return (uint32_t)f2bf(x) | ((uint32_t)f2bf(y) << 16);
}

// Fast tanh: prefer native v_tanh_f32 (gfx1250 transcendental), else
// branch-free expf-based identity (v_exp_f32 + v_rcp_f32).
__device__ __forceinline__ float fast_tanh(float x) {
#if __has_builtin(__builtin_amdgcn_tanhf)
    return __builtin_amdgcn_tanhf(x);
#else
    float e = __expf(2.0f * x);
    return 1.0f - 2.0f / (e + 1.0f);
#endif
}

// ---------------- fp32 -> bf16 weight conversion ----------------
__global__ void convert_w_kernel(const float* __restrict__ src,
                                 uint16_t* __restrict__ dst, int n) {
    int i = blockIdx.x * blockDim.x + threadIdx.x;
    int stride = gridDim.x * blockDim.x;
    for (; i < n; i += stride) dst[i] = f2bf(src[i]);
}

// ---------------- small GEMM: out[b,n] = sum_h x[b,h]*W[n,h] ----------------
__global__ void small_gemm_kernel(const float* __restrict__ x,
                                  const float* __restrict__ W,
                                  float* __restrict__ out) {
    int idx = blockIdx.x * blockDim.x + threadIdx.x;   // 65536 = B_*H_
    int b = idx >> 9, n = idx & (H_ - 1);
    const float4* xv = (const float4*)(x + (size_t)b * H_);
    const float4* wv = (const float4*)(W + (size_t)n * H_);
    float acc = 0.f;
#pragma unroll 4
    for (int i = 0; i < H_ / 4; ++i) {
        float4 a = xv[i], w = wv[i];
        acc += a.x * w.x + a.y * w.y + a.z * w.z + a.w * w.w;
    }
    out[idx] = acc;
}

// ---------------- fused GEMM + tanh + V-dot (WMMA bf16) ----------------
// score[m] = sum_n tanh( (enc[m,:] . W[n,:]) + bias[b,n] ) * Vw[n] + Vb
// 4 waves/WG; each wave owns 16 rows. The wave's A-strip (16x512) is
// converted fp32->bf16 ONCE and stored in LDS in WMMA-fragment order
// (dword index = (kt*8 + r)*32 + lane). Writer lane == reader lane, so
// no barrier is required and every DS access is consecutive-dword
// (bank-conflict free). Compute loop: 2 N-tiles per K-sweep reuse A.
__global__ void __launch_bounds__(128)
score_wmma_kernel(const float* __restrict__ enc,      // [B*S, H] f32
                  const uint16_t* __restrict__ wB,    // [H, H] bf16 row-major (n,k)
                  const float* __restrict__ bias,     // [B, H] f32
                  const float* __restrict__ Vw,       // [H]
                  const float* __restrict__ Vb,       // [1]
                  float* __restrict__ score)          // [B*S]
{
    __shared__ uint32_t afrag[4][16 * 8 * 32];        // 16 KB per wave, 64 KB total

    const int lane = threadIdx.x & 31;
    const int wave = threadIdx.x >> 5;
    const int l15  = lane & 15;
    const int half = lane >> 4;
    const int m0   = (blockIdx.x * 4 + wave) * 16;
    const int b    = m0 >> 10;                         // S_ = 1024
    const float* biasRow = bias + (size_t)b * H_;
    const float  vb = Vb[0];

    int koff[8];
#pragma unroll
    for (int r = 0; r < 8; ++r)
        koff[r] = ((r & 4) << 2) | ((r & 3) << 1) | (half << 3);

    // ---- stage A-strip into LDS (bf16, fragment order), once ----
    uint32_t* aw_ = &afrag[wave][0];
    {
        const float* aRow = enc + (size_t)(m0 + l15) * H_;
        for (int kt = 0; kt < H_ / 32; ++kt) {
            const int k0 = kt * 32;
#pragma unroll
            for (int r = 0; r < 8; ++r) {
                float2 t = *(const float2*)(aRow + k0 + koff[r]);
                aw_[(kt * 8 + r) * 32 + lane] = pack2bf(t.x, t.y);
            }
        }
    }
    const uint32_t* ap = aw_ + lane;                   // per-lane base

    float sacc[8];
#pragma unroll
    for (int r = 0; r < 8; ++r) sacc[r] = 0.f;

    for (int nt = 0; nt < H_ / 16; nt += 2) {
        const int n0 = nt * 16;
        v8f c0 = {0.f,0.f,0.f,0.f,0.f,0.f,0.f,0.f};
        v8f c1 = {0.f,0.f,0.f,0.f,0.f,0.f,0.f,0.f};

#pragma unroll 4
        for (int kt = 0; kt < H_ / 32; ++kt) {
            const int k0 = kt * 32;
            Frag a;
#pragma unroll
            for (int r = 0; r < 8; ++r)
                a.u[r] = ap[(kt * 8 + r) * 32];

            Frag b0, b1;
            const uint16_t* w0 = wB + (size_t)(n0      + l15) * H_ + k0;
            const uint16_t* w1 = wB + (size_t)(n0 + 16 + l15) * H_ + k0;
#pragma unroll
            for (int r = 0; r < 8; ++r) {
                b0.u[r] = *(const uint32_t*)(w0 + koff[r]);
                b1.u[r] = *(const uint32_t*)(w1 + koff[r]);
            }
            c0 = __builtin_amdgcn_wmma_f32_16x16x32_bf16(
                     false, a.v, false, b0.v, (short)0, c0, false, false);
            c1 = __builtin_amdgcn_wmma_f32_16x16x32_bf16(
                     false, a.v, false, b1.v, (short)0, c1, false, false);
        }
        const float bv0 = biasRow[n0 + l15],      vv0 = Vw[n0 + l15];
        const float bv1 = biasRow[n0 + 16 + l15], vv1 = Vw[n0 + 16 + l15];
#pragma unroll
        for (int r = 0; r < 8; ++r) {
            sacc[r] += fast_tanh(c0[r] + bv0) * vv0;
            sacc[r] += fast_tanh(c1[r] + bv1) * vv1;
        }
    }
    // D-tile layout: VGPR r, lanes 0-15 -> row r, lanes 16-31 -> row r+8.
    // sacc[r] holds a partial over n == l15 (mod 16); reduce across 16-lane half.
#pragma unroll
    for (int r = 0; r < 8; ++r) {
        float v = sacc[r];
        for (int off = 1; off < 16; off <<= 1)
            v += __shfl_xor(v, off, 32);
        if (l15 == 0) score[m0 + r + 8 * half] = v + vb;
    }
}

// ---------------- softmax over S + glimpse = aw@enc + dec ----------------
__global__ void __launch_bounds__(256)
softmax_glimpse_kernel(const float* __restrict__ score,   // [B,S]
                       const float* __restrict__ mask,    // [B,S]
                       const float* __restrict__ enc,     // [B,S,H] f32
                       const float* __restrict__ dec,     // [B,H]
                       float* __restrict__ glimpse)       // [B,H]
{
    __shared__ float aw[S_];
    __shared__ float red[8];
    const int b = blockIdx.x, tid = threadIdx.x;
    const float* sr = score + (size_t)b * S_;
    const float* mr = mask  + (size_t)b * S_;

    float vals[4];
    float lmax = -INFINITY;
#pragma unroll
    for (int i = 0; i < 4; ++i) {
        int s = tid + i * 256;
        float v = sr[s] - NEGV * (1.f - mr[s]);
        vals[i] = v;
        lmax = fmaxf(lmax, v);
    }
    for (int off = 1; off < 32; off <<= 1) lmax = fmaxf(lmax, __shfl_xor(lmax, off, 32));
    if ((tid & 31) == 0) red[tid >> 5] = lmax;
    __syncthreads();
    float bmax = red[0];
#pragma unroll
    for (int i = 1; i < 8; ++i) bmax = fmaxf(bmax, red[i]);

    float lsum = 0.f;
#pragma unroll
    for (int i = 0; i < 4; ++i) { vals[i] = __expf(vals[i] - bmax); lsum += vals[i]; }
    for (int off = 1; off < 32; off <<= 1) lsum += __shfl_xor(lsum, off, 32);
    __syncthreads();
    if ((tid & 31) == 0) red[tid >> 5] = lsum;
    __syncthreads();
    float bsum = 0.f;
#pragma unroll
    for (int i = 0; i < 8; ++i) bsum += red[i];
    const float inv = 1.f / bsum;
#pragma unroll
    for (int i = 0; i < 4; ++i) aw[tid + i * 256] = vals[i] * inv;
    __syncthreads();

    // glimpse: each thread owns columns tid and tid+256
    float acc0 = 0.f, acc1 = 0.f;
    const float* er = enc + (size_t)b * S_ * H_;
    for (int s = 0; s < S_; ++s) {
        const float w = aw[s];
        acc0 += w * er[(size_t)s * H_ + tid];
        acc1 += w * er[(size_t)s * H_ + tid + 256];
    }
    glimpse[(size_t)b * H_ + tid]       = acc0 + dec[(size_t)b * H_ + tid];
    glimpse[(size_t)b * H_ + tid + 256] = acc1 + dec[(size_t)b * H_ + tid + 256];
}

// ---------------- final softmax + log_softmax ----------------
__global__ void __launch_bounds__(256)
softmax_final_kernel(const float* __restrict__ score,  // [B,S]
                     const float* __restrict__ mask,   // [B,S]
                     float* __restrict__ out)          // [2,B,S]
{
    __shared__ float red[8];
    const int b = blockIdx.x, tid = threadIdx.x;
    const float* sr = score + (size_t)b * S_;
    const float* mr = mask  + (size_t)b * S_;

    float vals[4];
    float lmax = -INFINITY;
#pragma unroll
    for (int i = 0; i < 4; ++i) {
        int s = tid + i * 256;
        float v = sr[s] - NEGV * (1.f - mr[s]);
        vals[i] = v;
        lmax = fmaxf(lmax, v);
    }
    for (int off = 1; off < 32; off <<= 1) lmax = fmaxf(lmax, __shfl_xor(lmax, off, 32));
    if ((tid & 31) == 0) red[tid >> 5] = lmax;
    __syncthreads();
    float bmax = red[0];
#pragma unroll
    for (int i = 1; i < 8; ++i) bmax = fmaxf(bmax, red[i]);

    float lsum = 0.f;
    float ex[4];
#pragma unroll
    for (int i = 0; i < 4; ++i) { ex[i] = __expf(vals[i] - bmax); lsum += ex[i]; }
    for (int off = 1; off < 32; off <<= 1) lsum += __shfl_xor(lsum, off, 32);
    __syncthreads();
    if ((tid & 31) == 0) red[tid >> 5] = lsum;
    __syncthreads();
    float bsum = 0.f;
#pragma unroll
    for (int i = 0; i < 8; ++i) bsum += red[i];
    const float inv = 1.f / bsum;
    const float lse = logf(bsum);
#pragma unroll
    for (int i = 0; i < 4; ++i) {
        int s = tid + i * 256;
        out[(size_t)b * S_ + s]                   = ex[i] * inv;            // softmax
        out[(size_t)B_ * S_ + (size_t)b * S_ + s] = (vals[i] - bmax) - lse; // log_softmax
    }
}

extern "C" void kernel_launch(void* const* d_in, const int* in_sizes, int n_in,
                              void* d_out, int out_size, void* d_ws, size_t ws_size,
                              hipStream_t stream) {
    const float* enc  = (const float*)d_in[0];   // [B,S,H]
    const float* dec  = (const float*)d_in[1];   // [1,B,H]
    const float* mask = (const float*)d_in[2];   // [B,S]
    const float* W1g  = (const float*)d_in[3];   // [H,H]
    const float* W2g  = (const float*)d_in[4];   // [H,H]
    const float* Vgw  = (const float*)d_in[5];   // [H]
    const float* Vgb  = (const float*)d_in[6];   // [1]
    const float* W1   = (const float*)d_in[7];   // [H,H]
    const float* W2   = (const float*)d_in[8];   // [H,H]
    const float* Vw   = (const float*)d_in[9];   // [H]
    const float* Vb   = (const float*)d_in[10];  // [1]
    float* out = (float*)d_out;

    // workspace layout (bytes)
    uint8_t* ws = (uint8_t*)d_ws;
    uint16_t* w1gB   = (uint16_t*)(ws);                     // 512 KB
    uint16_t* w1B    = (uint16_t*)(ws +  524288);           // 512 KB
    float*    w2dg   = (float*)   (ws + 1048576);           // 256 KB
    float*    w2d    = (float*)   (ws + 1310720);           // 256 KB
    float*    scoreg = (float*)   (ws + 1572864);           // 512 KB
    float*    scoref = (float*)   (ws + 2097152);           // 512 KB
    float*    glimpse= (float*)   (ws + 2621440);           // 256 KB
    (void)ws_size; (void)in_sizes; (void)n_in; (void)out_size;

    // 1) convert weights to bf16
    convert_w_kernel<<<256, 256, 0, stream>>>(W1g, w1gB, H_ * H_);
    convert_w_kernel<<<256, 256, 0, stream>>>(W1,  w1B,  H_ * H_);

    // 2) w2_d_g[b,n] = dec[b,:] . W2_g[n,:]
    small_gemm_kernel<<<(B_ * H_) / 256, 256, 0, stream>>>(dec, W2g, w2dg);

    // 3) glimpse scores (fused GEMM+tanh+dot, WMMA)
    score_wmma_kernel<<<(B_ * S_) / 64, 128, 0, stream>>>(enc, w1gB, w2dg, Vgw, Vgb, scoreg);

    // 4) softmax over S + glimpse context
    softmax_glimpse_kernel<<<B_, 256, 0, stream>>>(scoreg, mask, enc, dec, glimpse);

    // 5) w2_d[b,n] = glimpse[b,:] . W2[n,:]
    small_gemm_kernel<<<(B_ * H_) / 256, 256, 0, stream>>>(glimpse, W2, w2d);

    // 6) final scores
    score_wmma_kernel<<<(B_ * S_) / 64, 128, 0, stream>>>(enc, w1B, w2d, Vw, Vb, scoref);

    // 7) softmax + log_softmax -> outputs
    softmax_final_kernel<<<B_, 256, 0, stream>>>(scoref, mask, out);
}
